// AggregatedAttention_13829794693776
// MI455X (gfx1250) — matrix operand order repaired and checked by reading, not compiled, not run on hardware
//
#include <hip/hip_runtime.h>
#include <hip/hip_bf16.h>
#include <math.h>

// Problem constants (from setup_inputs): B=4, H=W=64, N=4096, C=256,
// NUM_HEADS=8, HEAD_DIM=16 (16 sub-heads of 16 dims), WS=3 (9 taps).
#define DIMC        256
#define BATCH       4
#define IMH         64
#define IMW         64
#define NPOS        (IMH * IMW)        // 4096
#define BN          (BATCH * NPOS)     // 16384 rows
#define QKVCOLS     768                // 256 q + 256 k + 256 v
#define SCALE_Q     0.25f              // HEAD_DIM^-0.5
#define LAMBDA_INIT 0.8f
#define EPS_RMS     1e-5f

typedef __attribute__((ext_vector_type(16))) __bf16 v16bf;
typedef __attribute__((ext_vector_type(8)))  __bf16 v8bf;
typedef __attribute__((ext_vector_type(8)))  float  v8f;

// bit-level fallback: float -> bf16 RNE
__device__ __forceinline__ __bf16 f2bf(float f) {
    unsigned u = __builtin_bit_cast(unsigned, f);
    unsigned r = (u + 0x7FFFu + ((u >> 16) & 1u)) >> 16;
    unsigned short s = (unsigned short)r;
    return __builtin_bit_cast(__bf16, s);
}

// ---------------------------------------------------------------------------
// Kernel 0: one-shot f32 -> bf16 conversion (x and weights), so the GEMM hot
// loop contains no conversion VALU at all.  Uses v_cvt_pk_bf16_f32 if the
// builtin exists on this toolchain.
// ---------------------------------------------------------------------------
__global__ __launch_bounds__(256) void cvt_f32_bf16(
    const float* __restrict__ src, __bf16* __restrict__ dst, int npairs)
{
    const int i = blockIdx.x * blockDim.x + threadIdx.x;
    if (i >= npairs) return;
    const float a = src[2 * i], b = src[2 * i + 1];
#if __has_builtin(__builtin_amdgcn_cvt_pk_bf16_f32)
    auto p = __builtin_amdgcn_cvt_pk_bf16_f32(a, b);
    *(decltype(p)*)&dst[2 * i] = p;
#else
    dst[2 * i]     = f2bf(a);
    dst[2 * i + 1] = f2bf(b);
#endif
}

// ---------------------------------------------------------------------------
// Kernel 1: fused QKV projection.  Y[BN,768] = x[BN,256] @ [q_w;kv_w]^T + bias
// One block per 16-row tile; the 8 KB bf16 x-tile is staged to LDS ONCE via
// async global->LDS b128 copies (ASYNCcnt), then 4 waves sweep all 12 column
// groups: 12 x 8 v_wmma_f32_16x16x32_bf16 per wave (K = 256).
// Epilogue routes/scales into qbuf (x SCALE), kbuf, vbuf workspaces.
// ---------------------------------------------------------------------------
__global__ __launch_bounds__(128) void qkv_gemm_wmma(
    const __bf16* __restrict__ xbf,     // [BN, 256] bf16
    const __bf16* __restrict__ wbf,     // [768, 256] bf16 ([q_w;kv_w])
    const float* __restrict__ q_b, const float* __restrict__ kv_b,
    float* __restrict__ qbuf, float* __restrict__ kbuf, float* __restrict__ vbuf)
{
    __shared__ __bf16 xs[16][DIMC];            // 8 KB of the WGP's 320 KB LDS

    const int tile = blockIdx.x;               // 16-row tile of [BN, *]
    const int wave = threadIdx.x >> 5;         // 4 waves per block (wave32)
    const int lane = threadIdx.x & 31;
    const int half = lane >> 4;                // 0: lanes 0-15, 1: lanes 16-31
    const int l16  = lane & 15;

    // ---- async stage of the contiguous 8 KB x tile: global -> LDS ----
    {
        const unsigned lds0 = (unsigned)(uintptr_t)&xs[0][0];
        const unsigned long long g0 =
            (unsigned long long)(uintptr_t)(xbf + (size_t)tile * 16 * DIMC);
        #pragma unroll
        for (int c = 0; c < 4; ++c) {
            const unsigned chunk = threadIdx.x + c * 128;   // 512 x 16B chunks
            const unsigned            loff = lds0 + chunk * 16u;
            const unsigned long long  gadr = g0 + (unsigned long long)chunk * 16ull;
            asm volatile("global_load_async_to_lds_b128 %0, %1, off"
                         :: "v"(loff), "v"(gadr) : "memory");
        }
        asm volatile("s_wait_asynccnt 0" ::: "memory");
    }
    __syncthreads();

    // ---- sweep all 12 column groups of 64 (wave owns 16 columns each) ----
    for (int g = 0; g < 12; ++g) {
        const int j = g * 64 + wave * 16 + l16;     // this lane's output column
        const __bf16* wrow = wbf + (size_t)j * DIMC;
        if (g < 11) __builtin_prefetch(wrow + 64 * DIMC, 0, 1);  // next group

        v8f acc = {};
        #pragma unroll
        for (int k0 = 0; k0 < DIMC; k0 += 32) {
            // A fragment 16x32 bf16 (ISA 7.12.2):
            // lane<16 : row=l16, elems = K{k0..k0+7, k0+16..k0+23}
            // lane>=16: row=l16, elems = K{k0+8..+15, k0+24..+31}
            v8bf lo = *(const v8bf*)&xs[l16][k0 + half * 8];
            v8bf hi = *(const v8bf*)&xs[l16][k0 + 16 + half * 8];
            v16bf a = __builtin_shufflevector(lo, hi,
                        0, 1, 2, 3, 4, 5, 6, 7, 8, 9, 10, 11, 12, 13, 14, 15);

            // B fragment 32x16 bf16: column = l16, K order same as A fragment
            v8bf blo = *(const v8bf*)(wrow + k0 + half * 8);
            v8bf bhi = *(const v8bf*)(wrow + k0 + 16 + half * 8);
            v16bf bfr = __builtin_shufflevector(blo, bhi,
                        0, 1, 2, 3, 4, 5, 6, 7, 8, 9, 10, 11, 12, 13, 14, 15);

            acc = __builtin_amdgcn_wmma_f32_16x16x32_bf16(
                      /*neg_a=*/false, a, /*neg_b=*/false, bfr,
                      /*c_mod=*/(short)0, acc, /*reuse_a=*/false, /*reuse_b=*/false);
        }

        // bias + routing.  C/D layout: VGPR r -> row r + 8*half, column l16.
        const float bias = (j < 256) ? q_b[j] : kv_b[j - 256];
        #pragma unroll
        for (int r = 0; r < 8; ++r) {
            const int gr  = tile * 16 + r + half * 8;
            const float v = acc[r] + bias;
            if (j < 256)      qbuf[(size_t)gr * DIMC + j]         = v * SCALE_Q;
            else if (j < 512) kbuf[(size_t)gr * DIMC + (j - 256)] = v;
            else              vbuf[(size_t)gr * DIMC + (j - 512)] = v;
        }
    }
}

// ---------------------------------------------------------------------------
// Kernel 2: fused neighborhood attention.  One thread = (batch, position, head
// pair h in 0..7).  Two 9-tap logit sets (sub-heads 2h, 2h+1), masked softmax,
// differential mix with lambda, aggregation over v, RMS norm, and the
// depthwise 3x3 lepe conv (shares the same 9 v-taps).
// ---------------------------------------------------------------------------
__global__ __launch_bounds__(256) void attn_fused(
    const float* __restrict__ qbuf, const float* __restrict__ kbuf,
    const float* __restrict__ vbuf,
    const float* __restrict__ lq1, const float* __restrict__ lk1,
    const float* __restrict__ lq2, const float* __restrict__ lk2,
    const float* __restrict__ subln_w,
    const float* __restrict__ lepe_w, const float* __restrict__ lepe_b,
    float* __restrict__ out)
{
    const int gid = blockIdx.x * blockDim.x + threadIdx.x;  // BN*8 threads
    const int h   = gid & 7;            // head pair
    const int nn  = gid >> 3;           // global row 0..BN-1
    const int n   = nn & (NPOS - 1);
    const int yy0 = n >> 6;
    const int xx0 = n & (IMW - 1);

    // lambda = exp(lq1.lk1) - exp(lq2.lk2) + 0.8
    float d1 = 0.f, d2 = 0.f;
    #pragma unroll
    for (int d = 0; d < 16; ++d) { d1 += lq1[d] * lk1[d]; d2 += lq2[d] * lk2[d]; }
    const float lam = __expf(d1) - __expf(d2) + LAMBDA_INIT;

    // q for sub-heads 2h, 2h+1: channels h*32 .. h*32+31 (already scaled)
    float q[32];
    const float* qp = qbuf + (size_t)nn * DIMC + h * 32;
    #pragma unroll
    for (int e = 0; e < 32; ++e) q[e] = qp[e];

    float s1[9], s2[9];
    int   nb[9];
    #pragma unroll
    for (int l = 0; l < 9; ++l) {
        const int di = l / 3 - 1, dj = l % 3 - 1;
        const int yy = yy0 + di, xx = xx0 + dj;
        const bool ok = (yy >= 0) && (yy < IMH) && (xx >= 0) && (xx < IMW);
        nb[l] = ok ? (nn + di * IMW + dj) : -1;
        float a = -INFINITY, b = -INFINITY;
        if (ok) {
            const float* kp = kbuf + (size_t)nb[l] * DIMC + h * 32;
            a = 0.f; b = 0.f;
            #pragma unroll
            for (int e = 0; e < 16; ++e) { a += q[e] * kp[e]; b += q[16 + e] * kp[16 + e]; }
        }
        s1[l] = a; s2[l] = b;
    }

    // masked softmax over the 9 taps, both sub-heads
    float m1 = -INFINITY, m2 = -INFINITY;
    #pragma unroll
    for (int l = 0; l < 9; ++l) { m1 = fmaxf(m1, s1[l]); m2 = fmaxf(m2, s2[l]); }
    float t1 = 0.f, t2 = 0.f;
    float p1[9], p2[9];
    #pragma unroll
    for (int l = 0; l < 9; ++l) {
        p1[l] = (nb[l] >= 0) ? __expf(s1[l] - m1) : 0.f;
        p2[l] = (nb[l] >= 0) ? __expf(s2[l] - m2) : 0.f;
        t1 += p1[l]; t2 += p2[l];
    }
    const float i1 = 1.f / t1, i2 = 1.f / t2;
    float amix[9];
    #pragma unroll
    for (int l = 0; l < 9; ++l) amix[l] = p1[l] * i1 - lam * (p2[l] * i2);

    // aggregate v (32 channels of this head pair) + depthwise lepe conv,
    // sharing the same 9 neighbor loads of v.
    float acc[32], lep[32];
    #pragma unroll
    for (int e = 0; e < 32; ++e) { acc[e] = 0.f; lep[e] = 0.f; }
    #pragma unroll
    for (int l = 0; l < 9; ++l) {
        if (nb[l] < 0) continue;
        const float* vp = vbuf + (size_t)nb[l] * DIMC + h * 32;
        #pragma unroll
        for (int e = 0; e < 32; ++e) {
            const float vv = vp[e];
            acc[e] += amix[l] * vv;
            lep[e] += vv * lepe_w[(size_t)(h * 32 + e) * 9 + l];
        }
    }

    // RMS norm over the 32 channels, scale by subln_w * (1 - lambda_init)
    float ms = 0.f;
    #pragma unroll
    for (int e = 0; e < 32; ++e) ms += acc[e] * acc[e];
    const float rn = rsqrtf(ms * (1.f / 32.f) + EPS_RMS);

    float* op = out + (size_t)nn * DIMC + h * 32;
    #pragma unroll
    for (int e = 0; e < 32; ++e) {
        op[e] = acc[e] * rn * subln_w[e] * (1.f - LAMBDA_INIT)
              + lep[e] + lepe_b[h * 32 + e];
    }
}

// ---------------------------------------------------------------------------
extern "C" void kernel_launch(void* const* d_in, const int* in_sizes, int n_in,
                              void* d_out, int out_size, void* d_ws, size_t ws_size,
                              hipStream_t stream) {
    const float* x      = (const float*)d_in[0];
    const float* q_w    = (const float*)d_in[1];
    const float* q_b    = (const float*)d_in[2];
    const float* kv_w   = (const float*)d_in[3];
    const float* kv_b   = (const float*)d_in[4];
    const float* lq1    = (const float*)d_in[5];
    const float* lk1    = (const float*)d_in[6];
    const float* lq2    = (const float*)d_in[7];
    const float* lk2    = (const float*)d_in[8];
    const float* subln  = (const float*)d_in[9];
    const float* lepe_w = (const float*)d_in[10];
    const float* lepe_b = (const float*)d_in[11];
    // d_in[12] padding_mask (recomputed on device), d_in[13]/[14] H/W (constants)

    __bf16* xbf = (__bf16*)d_ws;                       // [BN, 256] bf16
    __bf16* wbf = xbf + (size_t)BN * DIMC;             // [768, 256] bf16
    float*  qbuf = (float*)(wbf + (size_t)QKVCOLS * DIMC);
    float*  kbuf = qbuf + (size_t)BN * DIMC;
    float*  vbuf = kbuf + (size_t)BN * DIMC;
    float*  out  = (float*)d_out;

    // Kernel 0: one-shot bf16 conversion of x, q_w, kv_w
    {
        const int xpairs = BN * DIMC / 2;              // 2,097,152
        cvt_f32_bf16<<<xpairs / 256, 256, 0, stream>>>(x, xbf, xpairs);
        const int qwpairs = 256 * DIMC / 2;            // 32,768
        cvt_f32_bf16<<<qwpairs / 256, 256, 0, stream>>>(q_w, wbf, qwpairs);
        const int kvpairs = 512 * DIMC / 2;            // 65,536
        cvt_f32_bf16<<<kvpairs / 256, 256, 0, stream>>>(
            kv_w, wbf + (size_t)256 * DIMC, kvpairs);
    }

    // Kernel 1: 1024 row-tiles, 4 waves/block, each wave sweeps 12 col groups
    qkv_gemm_wmma<<<BN / 16, 128, 0, stream>>>(xbf, wbf, q_b, kv_b,
                                               qbuf, kbuf, vbuf);

    // Kernel 2: one thread per (row, head-pair) = BN*8 threads
    attn_fused<<<(BN * 8) / 256, 256, 0, stream>>>(
        qbuf, kbuf, vbuf, lq1, lk1, lq2, lk2, subln, lepe_w, lepe_b, out);
}
